// TernaryLinear_446676598850
// MI455X (gfx1250) — 8-bit matrix operand order repaired and checked by reading, compiled, not run on hardware
//
#include <hip/hip_runtime.h>
#include <stdint.h>

typedef __attribute__((ext_vector_type(8))) int v8i;

#define M_TOKENS 8192
#define KDIM 4096
#define NDIM 11008

__device__ __forceinline__ int quant_pack4(float4 v, float s) {
    int a = max(-128, min(127, __float2int_rn(v.x * s)));
    int b = max(-128, min(127, __float2int_rn(v.y * s)));
    int c = max(-128, min(127, __float2int_rn(v.z * s)));
    int d = max(-128, min(127, __float2int_rn(v.w * s)));
    return (a & 0xFF) | ((b & 0xFF) << 8) | ((c & 0xFF) << 16) | (d << 24);
}

__global__ void __launch_bounds__(256)
quant_kernel(const float* __restrict__ x, int8_t* __restrict__ xq,
             float* __restrict__ xscale) {
    __shared__ float red[256];
    const int m = blockIdx.x;
    const int t = threadIdx.x;
    const float4* xr = (const float4*)(x + (size_t)m * KDIM);

    float4 v[4];
    float amax = 0.0f;
#pragma unroll
    for (int i = 0; i < 4; ++i) {
        v[i] = xr[i * 256 + t];
        amax = fmaxf(amax, fmaxf(fmaxf(fabsf(v[i].x), fabsf(v[i].y)),
                                 fmaxf(fabsf(v[i].z), fabsf(v[i].w))));
    }
    red[t] = amax;
    __syncthreads();
#pragma unroll
    for (int s = 128; s > 0; s >>= 1) {
        if (t < s) red[t] = fmaxf(red[t], red[t + s]);
        __syncthreads();
    }
    const float scale = 127.0f / fmaxf(red[0], 1e-5f);
    if (t == 0) xscale[m] = scale;

    const size_t mtile_base = (size_t)(m >> 4) * (KDIM / 64) * 1024;
    const int lslot = m & 15;
#pragma unroll
    for (int i = 0; i < 4; ++i) {
        const int e    = (i * 256 + t) * 4;
        const int tile = e >> 6;
        const int kk   = e & 63;
        const int hib  = (kk >> 4) & 1;
        const int byt  = ((kk >> 5) << 4) + (kk & 15);
        int8_t* dst = xq + mtile_base + (size_t)tile * 1024
                         + (lslot + hib * 16) * 32 + byt;
        *(int*)dst = quant_pack4(v[i], scale);
    }
}

__device__ __forceinline__ int pack_i8(int4 a) {
    return (a.x & 0xFF) | ((a.y & 0xFF) << 8) | ((a.z & 0xFF) << 16) | (a.w << 24);
}

__global__ void __launch_bounds__(256)
pack_w_kernel(const int4* __restrict__ wt, int4* __restrict__ w8) {
    const size_t idx = (size_t)blockIdx.x * 256 + threadIdx.x;
    const int4* src = wt + idx * 4;
    int4 a = src[0], b = src[1], c = src[2], d = src[3];
    int4 o;
    o.x = pack_i8(a); o.y = pack_i8(b); o.z = pack_i8(c); o.w = pack_i8(d);
    w8[idx] = o;
}

#define TILE_STRIDE 65536

__device__ __forceinline__ v8i load2x128(const int8_t* p, int off0, int off1) {
    int4 c0 = *(const int4*)(p + off0);
    int4 c1 = *(const int4*)(p + off1);
    v8i r;
    r[0] = c0.x; r[1] = c0.y; r[2] = c0.z; r[3] = c0.w;
    r[4] = c1.x; r[5] = c1.y; r[6] = c1.z; r[7] = c1.w;
    return r;
}

#define WMMA_IU8(A, B, C) \
    __builtin_amdgcn_wmma_i32_16x16x64_iu8(true, (A), true, (B), (C), false, false)

__global__ void __launch_bounds__(256)
gemm_kernel(const int8_t* __restrict__ xq, const int8_t* __restrict__ w8,
            const float* __restrict__ xscale, const float* __restrict__ wscale_p,
            const float* __restrict__ bias, float* __restrict__ out) {
    const int lane = threadIdx.x & 31;
    const int warp = threadIdx.x >> 5;
    const int row  = lane & 15;
    const int hi   = lane >> 4;
    const int wm   = warp >> 2;
    const int wn   = warp & 3;

    const int m0 = blockIdx.y * 128 + wm * 64;
    const int n0 = blockIdx.x * 256 + wn * 64;

    const int8_t* pa = xq + (size_t)(m0 >> 4) * (KDIM / 64) * 1024 + lane * 32;
    const int8_t* pb = w8 + (size_t)(n0 + row) * KDIM + hi * 16;

    v8i acc[4][4] = {};

    for (int k = 0; k < KDIM; k += 64) {
        v8i a0 = load2x128(pa, 0 * TILE_STRIDE, 0 * TILE_STRIDE + 16);
        v8i a1 = load2x128(pa, 1 * TILE_STRIDE, 1 * TILE_STRIDE + 16);
        v8i a2 = load2x128(pa, 2 * TILE_STRIDE, 2 * TILE_STRIDE + 16);
        v8i a3 = load2x128(pa, 3 * TILE_STRIDE, 3 * TILE_STRIDE + 16);
#pragma unroll
        for (int j = 0; j < 4; ++j) {
            v8i b = load2x128(pb, j * TILE_STRIDE, j * TILE_STRIDE + 32);
            acc[0][j] = WMMA_IU8(a0, b, acc[0][j]);
            acc[1][j] = WMMA_IU8(a1, b, acc[1][j]);
            acc[2][j] = WMMA_IU8(a2, b, acc[2][j]);
            acc[3][j] = WMMA_IU8(a3, b, acc[3][j]);
        }
        pa += 1024;
        pb += 64;
    }

    const float wsc = wscale_p[0];
#pragma unroll
    for (int i = 0; i < 4; ++i) {
        float minv[8];
#pragma unroll
        for (int r = 0; r < 8; ++r)
            minv[r] = wsc / xscale[m0 + i * 16 + hi * 8 + r];
#pragma unroll
        for (int j = 0; j < 4; ++j) {
            const float bf = bias[n0 + j * 16 + row];
#pragma unroll
            for (int r = 0; r < 8; ++r) {
                const size_t m = (size_t)(m0 + i * 16 + hi * 8 + r);
                out[m * NDIM + n0 + j * 16 + row] =
                    (float)acc[i][j][r] * minv[r] + bf;
            }
        }
    }
}

extern "C" void kernel_launch(void* const* d_in, const int* in_sizes, int n_in,
                              void* d_out, int out_size, void* d_ws, size_t ws_size,
                              hipStream_t stream) {
    const float* x      = (const float*)d_in[0];
    const int*   wt     = (const int*)d_in[1];
    const float* wscale = (const float*)d_in[2];
    const float* bias   = (const float*)d_in[3];
    float*       out    = (float*)d_out;

    uint8_t* ws = (uint8_t*)d_ws;
    int8_t* xq  = (int8_t*)ws;
    int8_t* w8  = (int8_t*)(ws + (size_t)M_TOKENS * KDIM);
    float*  xsc = (float*)(ws + (size_t)M_TOKENS * KDIM + (size_t)NDIM * KDIM);

    quant_kernel<<<M_TOKENS, 256, 0, stream>>>(x, xq, xsc);

    const int pack_threads = (NDIM * KDIM) / 16;
    pack_w_kernel<<<pack_threads / 256, 256, 0, stream>>>((const int4*)wt, (int4*)w8);

    dim3 grid(NDIM / 256, M_TOKENS / 128);
    gemm_kernel<<<grid, 256, 0, stream>>>(xq, w8, xsc, wscale, bias, out);
}
